// MaxPowerBraTSGNN_72670846649171
// MI455X (gfx1250) — matrix operand order repaired and checked by reading, not compiled, not run on hardware
//
#include <hip/hip_runtime.h>
#include <hip/hip_bf16.h>

typedef __attribute__((ext_vector_type(16))) __bf16 v16bf;
typedef __attribute__((ext_vector_type(8)))  float  v8f;

__device__ __forceinline__ unsigned short f2bf(float f) {
  unsigned int u = __float_as_uint(f);
  u += 0x7fffu + ((u >> 16) & 1u);  // round-to-nearest-even
  return (unsigned short)(u >> 16);
}
__device__ __forceinline__ float bf2f(unsigned short s) {
  return __uint_as_float(((unsigned int)s) << 16);
}

// ---------------------------------------------------------------- utilities
__global__ void zero_f32(float* __restrict__ p, int n) {
  int i = blockIdx.x * blockDim.x + threadIdx.x;
  if (i < n) p[i] = 0.0f;
}

__global__ void deg_kernel(const long long* __restrict__ dst, float* __restrict__ deg, int nedges) {
  int e = blockIdx.x * blockDim.x + threadIdx.x;
  if (e < nedges) atomicAdd(&deg[dst[e]], 1.0f);
}

// x (f32, [nrows,12]) -> zero-padded bf16 [mpad, 32]
__global__ void cast_x(const float* __restrict__ x, unsigned short* __restrict__ out,
                       int nrows, int mpad) {
  int i = blockIdx.x * blockDim.x + threadIdx.x;
  if (i >= (mpad << 5)) return;
  int row = i >> 5, k = i & 31;
  float t = (row < nrows && k < 12) ? x[row * 12 + k] : 0.0f;
  out[i] = f2bf(t);
}

// W (f32,[kreal,NOUT]) -> bf16, pre-swizzled into fragment-linear chunk order:
// dst[i], i = (((kc*NT + nt)*32 + l)*16 + e), where k = kc*32 + (l&16)/2 + (e&7) + 2*(e&8),
// n = nt*16 + (l&15).  Each 32-row k-chunk becomes one contiguous NOUT*32 block
// matching the documented B-fragment VGPR layout, so staging is a linear copy.
template <int NT, int LOG2NT>
__global__ void cast_weight_sw(const float* __restrict__ W, unsigned short* __restrict__ out,
                               int kreal, int kpad) {
  constexpr int NOUT = NT * 16;
  int i = blockIdx.x * blockDim.x + threadIdx.x;
  if (i >= kpad * NOUT) return;
  int e = i & 15;
  int l = (i >> 4) & 31;
  int nt = (i >> 9) & (NT - 1);
  int kc = i >> (9 + LOG2NT);
  int k = kc * 32 + ((l & 16) >> 1) + (e & 7) + ((e & 8) << 1);
  int n = nt * 16 + (l & 15);
  out[i] = f2bf(k < kreal ? W[k * NOUT + n] : 0.0f);
}

// scatter-add bf16 features into f32 accumulator (L2-resident).
// LOG2C = log2(K/8); shifts/masks instead of integer division.
template <int LOG2C>
__global__ void agg_scatter(const long long* __restrict__ src, const long long* __restrict__ dst,
                            const unsigned short* __restrict__ h, float* __restrict__ agg,
                            int nedges) {
  constexpr int K = 8 << LOG2C;
  int idx = blockIdx.x * blockDim.x + threadIdx.x;
  if (idx >= (nedges << LOG2C)) return;
  int e = idx >> LOG2C;
  int c = idx & ((1 << LOG2C) - 1);
  long long s = src[e], d = dst[e];
  uint4 q = *reinterpret_cast<const uint4*>(h + (size_t)s * K + c * 8);
  const unsigned short* qs = (const unsigned short*)&q;
  float* ap = agg + (size_t)d * K + c * 8;
#pragma unroll
  for (int j = 0; j < 8; ++j) atomicAdd(&ap[j], bf2f(qs[j]));
}

// agg / max(deg,1) -> bf16 (pad rows -> 0)
template <int LOG2K>
__global__ void mean_cast(const float* __restrict__ agg, const float* __restrict__ deg,
                          unsigned short* __restrict__ abf, int nrows, int mpad) {
  int i = blockIdx.x * blockDim.x + threadIdx.x;
  if (i >= (mpad << LOG2K)) return;
  int row = i >> LOG2K;
  float t = 0.0f;
  if (row < nrows) t = agg[i] / fmaxf(deg[row], 1.0f);
  abf[i] = f2bf(t);
}

// ------------------------------------------------ fused dual-GEMM + L2 norm
// out[M,NOUT] = L2rownorm( Abf@Wl + Xbf@Wr + bias ), plus column sum/sumsq atomics.
// Block: 256 thr = 8 waves; wave owns 16 rows x NT*16 cols.
// Weights arrive pre-swizzled chunk-linear; each chunk is DMA'd to LDS with
// async global->LDS loads (ASYNCcnt) double-buffered against the WMMAs.
template <int NT>
__global__ __launch_bounds__(256) void sage_gemm_l2_kernel(
    const unsigned short* __restrict__ Abf, int Ka,
    const unsigned short* __restrict__ Xbf, int Kx,
    const unsigned short* __restrict__ Wlsw,  // swizzled [Ka/32][NOUT*32] bf16
    const unsigned short* __restrict__ Wrsw,  // swizzled [Kx/32][NOUT*32] bf16
    const float* __restrict__ bias,
    float* __restrict__ outp,
    float* __restrict__ colsum, float* __restrict__ colsumsq,
    int nrows) {
  constexpr int NOUT = NT * 16;
  constexpr int CHUNK = NOUT * 32;                  // u16 elements per k-chunk
  constexpr int RINSTR = (CHUNK * 2) / (8 * 512);   // async b128 issues per wave
  __shared__ __align__(32) unsigned short Bs[2][CHUNK];

  const int lane = threadIdx.x & 31;
  const int wave = threadIdx.x >> 5;
  const int rowBase = blockIdx.x * 128 + wave * 16;
  const int mrow = lane & 15;              // A row within tile; also B/C column
  const int khalfA = (lane < 16) ? 0 : 8;  // ISA 16-bit A layout lane split
  const int rhalf = (lane < 16) ? 0 : 8;   // ISA 32-bit C layout row split

  const int nca = Ka >> 5, nct = nca + (Kx >> 5);

  // Per-wave A row base pointers, hoisted so the k-loop only bumps by t*32.
  const unsigned short* aBase0 = Abf + (size_t)(rowBase + mrow) * Ka + khalfA;
  const unsigned short* aBase1 = Xbf + (size_t)(rowBase + mrow) * Kx + khalfA;

  // Issue one chunk's worth of async global->LDS b128 copies (512B per issue).
  auto stage_async = [&](int buf, const unsigned short* gsrc) {
    const unsigned ldsBase = (unsigned)(uintptr_t)(&Bs[buf][0]);
    const unsigned long long gaddr = (unsigned long long)(uintptr_t)gsrc;
#pragma unroll
    for (int r = 0; r < RINSTR; ++r) {
      unsigned goff = ((unsigned)(wave * RINSTR + r) * 32u + (unsigned)lane) * 16u;
      asm volatile("global_load_async_to_lds_b128 %0, %1, %2"
                   :
                   : "v"(ldsBase + goff), "v"(goff), "s"(gaddr)
                   : "memory");
    }
  };
  auto chunk_src = [&](int t) -> const unsigned short* {
    return (t < nca) ? (Wlsw + (size_t)t * CHUNK)
                     : (Wrsw + (size_t)(t - nca) * CHUNK);
  };

  v8f c[NT];
#pragma unroll
  for (int nt = 0; nt < NT; ++nt)
#pragma unroll
    for (int v = 0; v < 8; ++v) c[nt][v] = 0.0f;

  stage_async(0, chunk_src(0));
  for (int t = 0; t < nct; ++t) {
    asm volatile("s_wait_asynccnt 0x0" ::: "memory");  // this wave's slices landed
    __syncthreads();                                   // all waves' slices landed
    if (t + 1 < nct) stage_async((t + 1) & 1, chunk_src(t + 1));  // overlap next

    // A fragment: lanes0-15 K {0..7,16..23}, lanes16-31 K {8..15,24..31}
    const unsigned short* ap =
        (t < nca) ? (aBase0 + t * 32) : (aBase1 + (t - nca) * 32);
    union { v16bf v; uint4 q[2]; } ua;
    ua.q[0] = *reinterpret_cast<const uint4*>(ap);
    ua.q[1] = *reinterpret_cast<const uint4*>(ap + 16);

    const unsigned short* bb = &Bs[t & 1][0];
#pragma unroll
    for (int nt = 0; nt < NT; ++nt) {
      const v16bf b = *reinterpret_cast<const v16bf*>(bb + (nt * 32 + lane) * 16);
      c[nt] = __builtin_amdgcn_wmma_f32_16x16x32_bf16(
          /*neg_a=*/false, ua.v, /*neg_b=*/false, b,
          /*c_mod=*/(short)0, c[nt], /*reuse_a=*/false, /*reuse_b=*/false);
    }
  }

  // bias + per-row L2 normalize (reduce across each 16-lane half = one row)
#pragma unroll
  for (int v = 0; v < 8; ++v) {
    const int row = rowBase + rhalf + v;
    const bool valid = row < nrows;
    float ss = 0.0f;
#pragma unroll
    for (int nt = 0; nt < NT; ++nt) {
      float t = c[nt][v] + bias[nt * 16 + mrow];
      t = valid ? t : 0.0f;
      c[nt][v] = t;
      ss += t * t;
    }
#pragma unroll
    for (int m = 1; m <= 8; m <<= 1) ss += __shfl_xor(ss, m, 32);
    const float inv = 1.0f / fmaxf(sqrtf(ss), 1e-12f);
#pragma unroll
    for (int nt = 0; nt < NT; ++nt) c[nt][v] *= inv;
  }

  // store + BN column stats (lane l and l^16 share a column -> combine halves)
#pragma unroll
  for (int nt = 0; nt < NT; ++nt) {
    const int col = nt * 16 + mrow;
    float s = 0.0f, s2 = 0.0f;
#pragma unroll
    for (int v = 0; v < 8; ++v) {
      const int row = rowBase + rhalf + v;
      const float t = c[nt][v];
      if (row < nrows) outp[(size_t)row * NOUT + col] = t;
      s += t;
      s2 += t * t;
    }
    s += __shfl_xor(s, 16, 32);
    s2 += __shfl_xor(s2, 16, 32);
    if (lane < 16) {
      atomicAdd(&colsum[col], s);
      atomicAdd(&colsumsq[col], s2);
    }
  }
}

__global__ void bn_finalize(const float* __restrict__ colsum, const float* __restrict__ colsumsq,
                            const float* __restrict__ g, const float* __restrict__ be,
                            float* __restrict__ scale, float* __restrict__ shift,
                            int nrows, int nout) {
  int cx = threadIdx.x;
  if (cx >= nout) return;
  float mean = colsum[cx] / (float)nrows;
  float var = colsumsq[cx] / (float)nrows - mean * mean;
  var = fmaxf(var, 0.0f);
  float sc = g[cx] * rsqrtf(var + 1e-5f);
  scale[cx] = sc;
  shift[cx] = be[cx] - mean * sc;
}

template <int LOG2W>
__global__ void bn_apply(const float* __restrict__ outp, const float* __restrict__ scale,
                         const float* __restrict__ shift, unsigned short* __restrict__ hout,
                         int nrows, int mpad, int relu) {
  int i = blockIdx.x * blockDim.x + threadIdx.x;
  if (i >= (mpad << LOG2W)) return;
  int row = i >> LOG2W, col = i & ((1 << LOG2W) - 1);
  float t = 0.0f;
  if (row < nrows) {
    t = outp[i] * scale[col] + shift[col];
    if (relu) t = fmaxf(t, 0.0f);
  }
  hout[i] = f2bf(t);  // pad rows -> 0 so next layer's unguarded A loads are safe
}

// ---------------------------------------------------------------- MLP head
__global__ __launch_bounds__(256) void head_kernel(
    const unsigned short* __restrict__ h, const float* __restrict__ cw1,
    const float* __restrict__ cb1, const float* __restrict__ cw2,
    const float* __restrict__ cb2, float* __restrict__ out, int nrows) {
  int node = blockIdx.x * 256 + threadIdx.x;
  if (node >= nrows) return;
  float acc[64];
#pragma unroll
  for (int j = 0; j < 64; ++j) acc[j] = cb1[j];
  const unsigned short* hp = h + (size_t)node * 128;
  for (int k = 0; k < 128; ++k) {
    float hk = bf2f(hp[k]);
#pragma unroll
    for (int j = 0; j < 64; ++j) acc[j] += hk * cw1[k * 64 + j];  // uniform scalar weight loads
  }
  float logit = cb2[0];
#pragma unroll
  for (int j = 0; j < 64; ++j) logit += fmaxf(acc[j], 0.0f) * cw2[j];
  out[node] = logit;
}

// ---------------------------------------------------------------- host side
namespace {
constexpr int NNODES = 100000;
constexpr int NEDGES = 800000;
constexpr int MPAD = 100096;  // multiple of 128

struct Bufs {
  float* deg;
  float* aggf;  // f32 accumulator, reused as pre-BN output
  unsigned short* abf;
  unsigned short* wlbf;
  unsigned short* wrbf;
  float* colsum;
  float* colsumsq;
  float* bnscale;
  float* bnshift;
};

// LOG2K = log2(padded input width), NT = NOUT/16
template <int LOG2K, int NT>
void run_layer(const unsigned short* hin, int kreal,
               const float* wl, const float* wr, const float* bl,
               const float* g, const float* be, bool relu,
               unsigned short* hout, const long long* src, const long long* dst,
               const Bufs& B, hipStream_t stream) {
  constexpr int KPAD = 1 << LOG2K;
  constexpr int NOUT = NT * 16;
  constexpr int LOG2NT = (NT == 16) ? 4 : 3;
  constexpr int LOG2W = (NT == 16) ? 8 : 7;
  const int aggN = MPAD * KPAD;
  zero_f32<<<(aggN + 255) / 256, 256, 0, stream>>>(B.aggf, aggN);
  agg_scatter<LOG2K - 3><<<((NEDGES << (LOG2K - 3)) + 255) / 256, 256, 0, stream>>>(
      src, dst, hin, B.aggf, NEDGES);
  mean_cast<LOG2K><<<(aggN + 255) / 256, 256, 0, stream>>>(B.aggf, B.deg, B.abf, NNODES, MPAD);
  const int wN = KPAD * NOUT;
  cast_weight_sw<NT, LOG2NT><<<(wN + 255) / 256, 256, 0, stream>>>(wl, B.wlbf, kreal, KPAD);
  cast_weight_sw<NT, LOG2NT><<<(wN + 255) / 256, 256, 0, stream>>>(wr, B.wrbf, kreal, KPAD);
  zero_f32<<<1, 256, 0, stream>>>(B.colsum, NOUT);
  zero_f32<<<1, 256, 0, stream>>>(B.colsumsq, NOUT);
  sage_gemm_l2_kernel<NT><<<MPAD / 128, 256, 0, stream>>>(
      B.abf, KPAD, hin, KPAD, B.wlbf, B.wrbf, bl, B.aggf, B.colsum, B.colsumsq, NNODES);
  bn_finalize<<<1, NOUT, 0, stream>>>(B.colsum, B.colsumsq, g, be, B.bnscale, B.bnshift, NNODES, NOUT);
  bn_apply<LOG2W><<<((MPAD << LOG2W) + 255) / 256, 256, 0, stream>>>(
      B.aggf, B.bnscale, B.bnshift, hout, NNODES, MPAD, relu ? 1 : 0);
}
}  // namespace

extern "C" void kernel_launch(void* const* d_in, const int* in_sizes, int n_in,
                              void* d_out, int out_size, void* d_ws, size_t ws_size,
                              hipStream_t stream) {
  (void)in_sizes; (void)n_in; (void)out_size; (void)ws_size;
  const float* x = (const float*)d_in[0];
  const long long* ei = (const long long*)d_in[1];  // int64 edge_index [2, E]
  const float* wl0 = (const float*)d_in[2];
  const float* bl0 = (const float*)d_in[3];
  const float* wr0 = (const float*)d_in[4];
  const float* g0 = (const float*)d_in[5];
  const float* be0 = (const float*)d_in[6];
  const float* wlm = (const float*)d_in[7];
  const float* blm = (const float*)d_in[8];
  const float* wrm = (const float*)d_in[9];
  const float* gm = (const float*)d_in[10];
  const float* bem = (const float*)d_in[11];
  const float* wl4 = (const float*)d_in[12];
  const float* bl4 = (const float*)d_in[13];
  const float* wr4 = (const float*)d_in[14];
  const float* g4 = (const float*)d_in[15];
  const float* be4 = (const float*)d_in[16];
  const float* cw1 = (const float*)d_in[17];
  const float* cb1 = (const float*)d_in[18];
  const float* cw2 = (const float*)d_in[19];
  const float* cb2 = (const float*)d_in[20];

  char* ws = (char*)d_ws;
  size_t off = 0;
  auto alloc = [&](size_t bytes) -> void* {
    void* p = ws + off;
    off = (off + bytes + 255) & ~(size_t)255;
    return p;
  };
  Bufs B;
  B.deg = (float*)alloc((size_t)MPAD * 4);
  B.aggf = (float*)alloc((size_t)MPAD * 256 * 4);
  B.abf = (unsigned short*)alloc((size_t)MPAD * 256 * 2);
  unsigned short* hA = (unsigned short*)alloc((size_t)MPAD * 256 * 2);
  unsigned short* hB = (unsigned short*)alloc((size_t)MPAD * 256 * 2);
  B.wlbf = (unsigned short*)alloc(256 * 256 * 2);
  B.wrbf = (unsigned short*)alloc(256 * 256 * 2);
  B.colsum = (float*)alloc(256 * 4);
  B.colsumsq = (float*)alloc(256 * 4);
  B.bnscale = (float*)alloc(256 * 4);
  B.bnshift = (float*)alloc(256 * 4);

  const long long* src = ei;
  const long long* dst = ei + NEDGES;

  // degree (graph-constant across layers)
  zero_f32<<<(MPAD + 255) / 256, 256, 0, stream>>>(B.deg, MPAD);
  deg_kernel<<<(NEDGES + 255) / 256, 256, 0, stream>>>(dst, B.deg, NEDGES);

  // input features -> bf16, K padded 12 -> 32
  cast_x<<<((MPAD * 32) + 255) / 256, 256, 0, stream>>>(x, hA, NNODES, MPAD);

  // layer 0: 12(->32) -> 256, ReLU
  run_layer<5, 16>(hA, 12, wl0, wr0, bl0, g0, be0, true, hB, src, dst, B, stream);
  unsigned short* cur = hB;
  unsigned short* nxt = hA;
  // layers 1..3: 256 -> 256, ReLU
  for (int i = 0; i < 3; ++i) {
    run_layer<8, 16>(cur, 256, wlm + (size_t)i * 256 * 256, wrm + (size_t)i * 256 * 256,
                     blm + i * 256, gm + i * 256, bem + i * 256, true, nxt, src, dst, B, stream);
    unsigned short* t = cur; cur = nxt; nxt = t;
  }
  // layer 4: 256 -> 128, no ReLU
  run_layer<8, 8>(cur, 256, wl4, wr4, bl4, g4, be4, false, nxt, src, dst, B, stream);
  // classifier head -> logits f32 [N]
  head_kernel<<<(NNODES + 255) / 256, 256, 0, stream>>>(nxt, cw1, cb1, cw2, cb2, (float*)d_out, NNODES);
}
